// LinearSelfAttention_86045374808452
// MI455X (gfx1250) — matrix operand order repaired and checked by reading, not compiled, not run on hardware
//
#include <hip/hip_runtime.h>
#include <hip/hip_bf16.h>

#define BATCH 2
#define TCTX  2048
#define EMB   1024
#define NHEAD 16
#define HEAD  64

typedef __attribute__((ext_vector_type(16))) __bf16 bf16x16;
typedef __attribute__((ext_vector_type(8)))  __bf16 bf16x8;
typedef __attribute__((ext_vector_type(8)))  float  f32x8;
typedef __attribute__((ext_vector_type(4)))  unsigned int u32x4;
typedef __attribute__((ext_vector_type(8)))  int          i32x8;
typedef __attribute__((ext_vector_type(4)))  int          i32x4;

#if defined(__has_builtin)
#if __has_builtin(__builtin_amdgcn_tensor_load_to_lds) && __has_builtin(__builtin_amdgcn_s_wait_tensorcnt)
#define HAVE_TDM 1
#endif
#endif

__device__ __forceinline__ __bf16 f2bf(float f) { return (__bf16)f; }
__device__ __forceinline__ float  bf2f(__bf16 h) { return (float)h; }

__device__ __forceinline__ bf16x16 cat8(bf16x8 lo, bf16x8 hi) {
  return __builtin_shufflevector(lo, hi, 0,1,2,3,4,5,6,7,8,9,10,11,12,13,14,15);
}

// 16-bit A-matrix 16x32 fragment: lanes 0-15 hold K=kb..kb+15 with kb=0,
// lanes 16-31 with kb=16 (contiguous, matches B). Caller passes rowptr = row base + kb.
__device__ __forceinline__ bf16x16 load_a_frag(const __bf16* rowptr) {
  return *(const bf16x16*)(rowptr);
}

__device__ __forceinline__ f32x8 wmma_bf16(bf16x16 a, bf16x16 b, f32x8 c) {
  return __builtin_amdgcn_wmma_f32_16x16x32_bf16(false, a, false, b, (short)0, c, false, false);
}

#ifdef HAVE_TDM
// Issue one TDM 2-D tile load (bf16 elements) into LDS per CDNA5 ISA §8 D# layout.
// Tensor dims == tile dims (caller guarantees in-bounds), stride in elements.
__device__ __forceinline__ void tdm_load_2d(unsigned int lds_off, const void* gaddr,
                                            unsigned int tile_w, unsigned int tile_h,
                                            unsigned long long stride_elems) {
  const unsigned long long ga = (unsigned long long)(uintptr_t)gaddr;
  u32x4 g0;
  g0[0] = 1u;                                             // count=1, user-mode, no gather
  g0[1] = lds_off;                                        // lds_addr [63:32]
  g0[2] = (unsigned int)ga;                               // global_addr[31:0]
  g0[3] = (unsigned int)((ga >> 32) & 0x1FFFFFFull) | (2u << 30);  // ga[56:32] | type=2

  // group1 bit layout (ISA §8.4): data_size@16 (1 => 2B), tensor_dim0@48 (32b),
  // tensor_dim1@80 (32b), tile_dim0@112 (16b), tile_dim1@128 (16b), tile_dim2@144 (16b),
  // tensor_dim0_stride@160 (48b), tensor_dim1_stride@208 (48b).
  const unsigned long long q0 = (1ull << 16) | ((unsigned long long)(tile_w & 0xFFFFu) << 48);
  const unsigned long long q1 = (unsigned long long)(tile_w >> 16)
                              | ((unsigned long long)tile_h << 16)
                              | ((unsigned long long)tile_w << 48);
  const unsigned long long q2 = (unsigned long long)tile_h
                              | ((stride_elems & 0xFFFFFFFFull) << 32);
  const unsigned long long q3 = (stride_elems >> 32) & 0xFFFFull;   // dim1_stride unused (2-D)

  i32x8 g1;
  g1[0] = (int)(unsigned int)q0;  g1[1] = (int)(unsigned int)(q0 >> 32);
  g1[2] = (int)(unsigned int)q1;  g1[3] = (int)(unsigned int)(q1 >> 32);
  g1[4] = (int)(unsigned int)q2;  g1[5] = (int)(unsigned int)(q2 >> 32);
  g1[6] = (int)(unsigned int)q3;  g1[7] = (int)(unsigned int)(q3 >> 32);

  i32x4 z4 = {0, 0, 0, 0};
#if __clang_major__ >= 23
  i32x8 z8 = {0, 0, 0, 0, 0, 0, 0, 0};
  __builtin_amdgcn_tensor_load_to_lds(g0, g1, z4, z4, z8, 0);
#else
  __builtin_amdgcn_tensor_load_to_lds(g0, g1, z4, z4, 0);
#endif
}
#endif  // HAVE_TDM

// ---------------- layernorm (parameter-free) -> bf16 ----------------
__global__ __launch_bounds__(256) void normalize_to_bf16(const float* __restrict__ x,
                                                         __bf16* __restrict__ xn) {
  __shared__ float s1[256], s2[256];
  const int rowid = blockIdx.x;
  const float* xr = x + (size_t)rowid * EMB;
  float v[4]; float sum = 0.f, sq = 0.f;
  #pragma unroll
  for (int i = 0; i < 4; ++i) {
    v[i] = xr[threadIdx.x + i * 256];
    sum += v[i]; sq += v[i] * v[i];
  }
  s1[threadIdx.x] = sum; s2[threadIdx.x] = sq;
  __syncthreads();
  for (int s = 128; s > 0; s >>= 1) {
    if ((int)threadIdx.x < s) { s1[threadIdx.x] += s1[threadIdx.x + s]; s2[threadIdx.x] += s2[threadIdx.x + s]; }
    __syncthreads();
  }
  const float mu = s1[0] * (1.f / EMB);
  const float var = s2[0] * (1.f / EMB) - mu * mu;
  const float rstd = rsqrtf(var + 1e-5f);
  __bf16* outr = xn + (size_t)rowid * EMB;
  #pragma unroll
  for (int i = 0; i < 4; ++i) outr[threadIdx.x + i * 256] = f2bf((v[i] - mu) * rstd);
}

// ---------------- fp32 -> bf16 weight convert ----------------
__global__ __launch_bounds__(256) void f32_to_bf16(const float* __restrict__ in,
                                                   __bf16* __restrict__ out, int n) {
  int i = blockIdx.x * 256 + threadIdx.x;
  if (i < n) out[i] = f2bf(in[i]);
}

// ---------------- TN GEMM: C[M,N] = A[M,K] * B[N,K]^T (bf16 in, f32 acc) ----------------
template <bool OUT_BF16>
__global__ __launch_bounds__(256) void gemm_tn(const __bf16* __restrict__ A,
                                               const __bf16* __restrict__ B,
                                               void* __restrict__ Cout,
                                               int M, int N, int K) {
  const int lane = threadIdx.x & 31;
  const int wave = threadIdx.x >> 5;           // 8 waves
  const int row  = lane & 15;
  const int hi   = lane >> 4;
  const int mblk = blockIdx.y * 128 + (wave & 3) * 32;   // wave: 32 rows
  const int nblk = blockIdx.x * 128 + (wave >> 2) * 64;  // wave: 64 cols
  const int kbA  = hi ? 16 : 0;  // A-fragment per-lane K base
  const int kbB  = hi ? 16 : 0;  // B-fragment per-lane K base (32x16 layout)

  f32x8 acc[2][4] = {};

  const __bf16* Arow0 = A + (size_t)(mblk + row) * K + kbA;
  const __bf16* Arow1 = A + (size_t)(mblk + 16 + row) * K + kbA;
  const __bf16* Brow[4];
  #pragma unroll
  for (int j = 0; j < 4; ++j) Brow[j] = B + (size_t)(nblk + j * 16 + row) * K + kbB;

  for (int k0 = 0; k0 < K; k0 += 32) {
    bf16x16 a0 = load_a_frag(Arow0 + k0);
    bf16x16 a1 = load_a_frag(Arow1 + k0);
    #pragma unroll
    for (int j = 0; j < 4; ++j) {
      bf16x16 b = *(const bf16x16*)(Brow[j] + k0);
      acc[0][j] = wmma_bf16(a0, b, acc[0][j]);
      acc[1][j] = wmma_bf16(a1, b, acc[1][j]);
    }
  }
  // epilogue: C/D layout -> VGPR r holds (m = r + 8*hi, n = lane&15)
  #pragma unroll
  for (int i = 0; i < 2; ++i) {
    const int mbase = mblk + i * 16 + hi * 8;
    #pragma unroll
    for (int j = 0; j < 4; ++j) {
      const int n = nblk + j * 16 + row;
      #pragma unroll
      for (int r = 0; r < 8; ++r) {
        const size_t off = (size_t)(mbase + r) * N + n;
        if constexpr (OUT_BF16) ((__bf16*)Cout)[off] = f2bf(acc[i][j][r]);
        else                    ((float*)Cout)[off]  = acc[i][j][r];
      }
    }
  }
}

// ---------------- split heads + elu(q,k)+1, v transposed ----------------
__global__ __launch_bounds__(256) void split_heads(const __bf16* __restrict__ qkv,
                                                   __bf16* __restrict__ q,
                                                   __bf16* __restrict__ k,
                                                   __bf16* __restrict__ vT) {
  const size_t idx = (size_t)blockIdx.x * 256 + threadIdx.x;  // over B*T*3C
  const int c3 = (int)(idx % (3 * EMB));
  const int bt = (int)(idx / (3 * EMB));
  const int part = c3 / EMB;
  const int within = c3 % EMB;
  const int h = within / HEAD, d = within % HEAD;
  const int b = bt / TCTX, t = bt % TCTX;
  float val = bf2f(qkv[idx]);
  const int bh = b * NHEAD + h;
  if (part == 0) {
    val = (val > 0.f) ? (val + 1.f) : __expf(val);          // elu(x)+1
    q[((size_t)bh * TCTX + t) * HEAD + d] = f2bf(val);
  } else if (part == 1) {
    val = (val > 0.f) ? (val + 1.f) : __expf(val);
    k[((size_t)bh * TCTX + t) * HEAD + d] = f2bf(val);
  } else {
    vT[((size_t)bh * HEAD + d) * TCTX + t] = f2bf(val);     // transposed for B-frag loads
  }
}

// ---------------- causal linear attention, flash-style, WMMA + TDM ----------------
// grid: (TCTX/64, BATCH*NHEAD), block: 128 (4 waves). Wave w owns query rows t0..t0+15.
// K / V^T 64x64 tiles are DMA'd into LDS by the Tensor Data Mover (double-buffered);
// waves consume them as WMMA B-fragments via ds loads.
__global__ __launch_bounds__(128) void linattn(const __bf16* __restrict__ q,
                                               const __bf16* __restrict__ k,
                                               const __bf16* __restrict__ vT,
                                               __bf16* __restrict__ y) {
  __shared__ __align__(64) __bf16 sK[2][64 * 64];   // K tile  [s][d]
  __shared__ __align__(64) __bf16 sV[2][64 * 64];   // V^T tile [d][s]
  __shared__ __align__(64) __bf16 sS[4][16 * 64];   // per-wave S strip (D-layout -> A-layout)

  const int tid  = threadIdx.x;
  const int lane = tid & 31;
  const int wave = tid >> 5;
  const int row  = lane & 15;
  const int hi   = lane >> 4;
  const int qblk = blockIdx.x;
  const int bh   = blockIdx.y;
  const int t0   = qblk * 64 + wave * 16;

  const __bf16* qbase = q  + (size_t)bh * TCTX * HEAD;
  const __bf16* kbase = k  + (size_t)bh * TCTX * HEAD;
  const __bf16* vbase = vT + (size_t)bh * HEAD * TCTX;

  const int kbA = hi ? 16 : 0;
  const int kbB = hi ? 16 : 0;

  // Q fragments (persist all iterations): contraction over d (64 -> two 16x32 frags)
  const __bf16* qrow = qbase + (size_t)(t0 + row) * HEAD;
  const bf16x16 aq0 = load_a_frag(qrow + kbA);
  const bf16x16 aq1 = load_a_frag(qrow + 32 + kbA);

  f32x8 yacc[4] = {};   // 4 d-subtiles of 16
  f32x8 dsum = {};      // per-lane partial row sums (masked S)

  __bf16* sw = &sS[wave][0];

  // Touch the staged LDS buffers so the compiler cannot treat them as store-free
  // (the TDM writes are invisible to the IR).
  ((unsigned int*)&sK[0][0])[tid] = 0u; ((unsigned int*)&sK[1][0])[tid] = 0u;
  ((unsigned int*)&sV[0][0])[tid] = 0u; ((unsigned int*)&sV[1][0])[tid] = 0u;
  __syncthreads();

#ifdef HAVE_TDM
  if (wave == 0) {  // uniform branch; TDM ignores EXEC, one issue per workgroup
    tdm_load_2d((unsigned int)(uintptr_t)&sK[0][0], kbase, HEAD, 64, HEAD);
    tdm_load_2d((unsigned int)(uintptr_t)&sV[0][0], vbase, 64, 64, TCTX);
  }
#endif

  for (int j = 0; j <= qblk; ++j) {
    const int buf   = j & 1;
    const int sbase = j * 64;

#ifdef HAVE_TDM
    if (wave == 0) {
      if (j < qblk) {  // prefetch next tile pair into the other buffer, overlap with compute
        const int nb = buf ^ 1;
        tdm_load_2d((unsigned int)(uintptr_t)&sK[nb][0], kbase + (size_t)(sbase + 64) * HEAD,
                    HEAD, 64, HEAD);
        tdm_load_2d((unsigned int)(uintptr_t)&sV[nb][0], vbase + (sbase + 64),
                    64, 64, TCTX);
        __builtin_amdgcn_s_wait_tensorcnt(2);   // retire current pair; next pair in flight
      } else {
        __builtin_amdgcn_s_wait_tensorcnt(0);
      }
    }
#else
    {  // fallback: cooperative global -> LDS staging of the current tile pair
      bf16x8* dK = (bf16x8*)&sK[buf][0];
      bf16x8* dV = (bf16x8*)&sV[buf][0];
      const __bf16* gK = kbase + (size_t)sbase * HEAD;   // contiguous 64*64
      #pragma unroll
      for (int t = 0; t < 4; ++t) {
        const int c = tid + t * 128;                     // 512 chunks of 8
        dK[c] = *(const bf16x8*)(gK + c * 8);
        const int r = c >> 3, c8 = c & 7;                // V^T rows stride TCTX
        dV[c] = *(const bf16x8*)(vbase + (size_t)r * TCTX + sbase + c8 * 8);
      }
    }
#endif
    __syncthreads();   // tiles visible to all waves

    // ---- S = Q K^T over 4 key subtiles (B-frags from LDS) ----
    #pragma unroll
    for (int sub = 0; sub < 4; ++sub) {
      const __bf16* krow = &sK[buf][(sub * 16 + row) * 64];
      bf16x16 bk0 = *(const bf16x16*)(krow + kbB);          // d = 0..31
      bf16x16 bk1 = *(const bf16x16*)(krow + 32 + kbB);     // d = 32..63
      f32x8 s = {};
      s = wmma_bf16(aq0, bk0, s);
      s = wmma_bf16(aq1, bk1, s);
      if (j == qblk) {                                      // causal mask on diagonal block
        const int scol = sbase + sub * 16 + row;
        #pragma unroll
        for (int r = 0; r < 8; ++r) {
          const int t = t0 + r + hi * 8;
          if (scol > t) s[r] = 0.f;
        }
      }
      dsum += s;
      #pragma unroll
      for (int r = 0; r < 8; ++r)                           // stash S as bf16 in LDS strip
        sw[(r + hi * 8) * 64 + sub * 16 + row] = f2bf(s[r]);
    }
    // ---- Y += S V : re-load S as A-fragments (contraction over s) ----
    const __bf16* srowp = sw + row * 64;
    bf16x16 as0 = load_a_frag(srowp + kbA);                 // s = 0..31
    bf16x16 as1 = load_a_frag(srowp + 32 + kbA);            // s = 32..63
    #pragma unroll
    for (int dsub = 0; dsub < 4; ++dsub) {
      const __bf16* vrow = &sV[buf][(dsub * 16 + row) * 64];
      bf16x16 bv0 = *(const bf16x16*)(vrow + kbB);
      bf16x16 bv1 = *(const bf16x16*)(vrow + 32 + kbB);
      yacc[dsub] = wmma_bf16(as0, bv0, yacc[dsub]);
      yacc[dsub] = wmma_bf16(as1, bv1, yacc[dsub]);
    }
    __syncthreads();   // all waves done reading buf before it is overwritten
  }

  // finish row sums: reduce over the 16 lanes of each half-wave (n = 0..15)
  f32x8 rs = dsum;
  #pragma unroll
  for (int m = 1; m < 16; m <<= 1) {
    #pragma unroll
    for (int r = 0; r < 8; ++r) rs[r] += __shfl_xor(rs[r], m, 32);
  }

  const int b = bh / NHEAD, h = bh % NHEAD;
  #pragma unroll
  for (int dsub = 0; dsub < 4; ++dsub) {
    #pragma unroll
    for (int r = 0; r < 8; ++r) {
      const int t = t0 + r + hi * 8;
      const float val = yacc[dsub][r] / rs[r];
      y[((size_t)(b * TCTX + t)) * EMB + h * HEAD + dsub * 16 + row] = f2bf(val);
    }
  }
}

extern "C" void kernel_launch(void* const* d_in, const int* in_sizes, int n_in,
                              void* d_out, int out_size, void* d_ws, size_t ws_size,
                              hipStream_t stream) {
  const float* x  = (const float*)d_in[0];
  const float* W1 = (const float*)d_in[1];
  const float* W2 = (const float*)d_in[2];

  char* ws = (char*)d_ws;
  size_t off = 0;
  auto alloc = [&](size_t bytes) -> void* {
    void* p = ws + off;
    off += (bytes + 255) & ~(size_t)255;
    return p;
  };
  __bf16* xn   = (__bf16*)alloc((size_t)BATCH * TCTX * EMB * 2);
  __bf16* w1b  = (__bf16*)alloc((size_t)3 * EMB * EMB * 2);
  __bf16* w2b  = (__bf16*)alloc((size_t)EMB * EMB * 2);
  __bf16* qkvb = (__bf16*)alloc((size_t)BATCH * TCTX * 3 * EMB * 2);
  __bf16* qb   = (__bf16*)alloc((size_t)BATCH * TCTX * EMB * 2);
  __bf16* kb   = (__bf16*)alloc((size_t)BATCH * TCTX * EMB * 2);
  __bf16* vtb  = (__bf16*)alloc((size_t)BATCH * TCTX * EMB * 2);
  __bf16* yb   = (__bf16*)alloc((size_t)BATCH * TCTX * EMB * 2);
  (void)ws_size; (void)in_sizes; (void)n_in; (void)out_size;

  normalize_to_bf16<<<BATCH * TCTX, 256, 0, stream>>>(x, xn);
  {
    const int n1 = 3 * EMB * EMB;
    f32_to_bf16<<<(n1 + 255) / 256, 256, 0, stream>>>(W1, w1b, n1);
    const int n2 = EMB * EMB;
    f32_to_bf16<<<(n2 + 255) / 256, 256, 0, stream>>>(W2, w2b, n2);
  }
  // qkv = xn @ W1^T : [4096,1024] x [3072,1024]^T
  gemm_tn<true><<<dim3(3 * EMB / 128, BATCH * TCTX / 128), 256, 0, stream>>>(
      xn, w1b, (void*)qkvb, BATCH * TCTX, 3 * EMB, EMB);
  {
    const size_t n = (size_t)BATCH * TCTX * 3 * EMB;
    split_heads<<<(unsigned)(n / 256), 256, 0, stream>>>(qkvb, qb, kb, vtb);
  }
  linattn<<<dim3(TCTX / 64, BATCH * NHEAD), 128, 0, stream>>>(qb, kb, vtb, yb);
  // out = y @ W2^T : [4096,1024] x [1024,1024]^T -> fp32 d_out
  gemm_tn<false><<<dim3(EMB / 128, BATCH * TCTX / 128), 256, 0, stream>>>(
      yb, w2b, d_out, BATCH * TCTX, EMB, EMB);
}
